// RNN_62938450756205
// MI455X (gfx1250) — compile-verified, matrix operand order checked by reading, not run on hardware
//
#include <hip/hip_runtime.h>
#include <hip/hip_bf16.h>
#include <math.h>

typedef __attribute__((ext_vector_type(16))) _Float16 v16h;
typedef __attribute__((ext_vector_type(8)))  _Float16 v8h;
typedef __attribute__((ext_vector_type(8)))  float    v8f;

#define SEQ      512
#define BATCH    1024
#define IN_DIM   64
#define HID      256
#define MTILE    16
#define HSTRIDE  264      // halfs per h-row (256 + 8 pad -> 528B, skews LDS banks, 16B aligned)
#define HBUFSZ   (MTILE * HSTRIDE)

__device__ __forceinline__ v8f wmma_f16(v16h a, v16h b, v8f c) {
    return __builtin_amdgcn_wmma_f32_16x16x32_f16(false, a, false, b, (short)0, c, false, false);
}

// A-fragment (16x32 f16) from an LDS hidden-state buffer.
// lane L: row = L%16, g = L/16; elems 0..7 -> K = 32kc+8g+e ; elems 8..15 -> K = 32kc+16+8g+(e-8)
__device__ __forceinline__ v16h afrag_lds(const _Float16* hbuf, int row, int kc, int g) {
    const _Float16* p = hbuf + row * HSTRIDE + 32 * kc + 8 * g;
    v8h lo = *(const v8h*)p;
    v8h hi = *(const v8h*)(p + 16);
    v16h a;
#pragma unroll
    for (int e = 0; e < 8; ++e) { a[e] = lo[e]; a[e + 8] = hi[e]; }
    return a;
}

// B-fragment (32x16 f16) from the pre-swizzled LDS weight buffer (32B contiguous per lane).
__device__ __forceinline__ v16h bfrag_lds(const _Float16* Wsw, int nt, int kc, int lane) {
    const _Float16* p = Wsw + (((nt * 8 + kc) * 32) + lane) * 16;
    v8h lo = *(const v8h*)p;
    v8h hi = *(const v8h*)(p + 8);
    v16h b;
#pragma unroll
    for (int e = 0; e < 8; ++e) { b[e] = lo[e]; b[e + 8] = hi[e]; }
    return b;
}

__device__ __forceinline__ v16h afrag_from_x(const float4 f0, const float4 f1,
                                             const float4 f2, const float4 f3) {
    v16h a;
    a[0]  = (_Float16)f0.x; a[1]  = (_Float16)f0.y; a[2]  = (_Float16)f0.z; a[3]  = (_Float16)f0.w;
    a[4]  = (_Float16)f1.x; a[5]  = (_Float16)f1.y; a[6]  = (_Float16)f1.z; a[7]  = (_Float16)f1.w;
    a[8]  = (_Float16)f2.x; a[9]  = (_Float16)f2.y; a[10] = (_Float16)f2.z; a[11] = (_Float16)f2.w;
    a[12] = (_Float16)f3.x; a[13] = (_Float16)f3.y; a[14] = (_Float16)f3.z; a[15] = (_Float16)f3.w;
    return a;
}

__global__ __launch_bounds__(256) void rnn_fused(
    const float* __restrict__ x,     const float* __restrict__ hidden,
    const float* __restrict__ Wih0,  const float* __restrict__ Whh0,
    const float* __restrict__ bih0,  const float* __restrict__ bhh0,
    const float* __restrict__ Wih1,  const float* __restrict__ Whh1,
    const float* __restrict__ bih1,  const float* __restrict__ bhh1,
    const float* __restrict__ Wfc,   const float* __restrict__ bfc,
    float* __restrict__ out)
{
    __shared__ _Float16 Wsw[16 * 8 * 32 * 16];   // swizzled W_ih1 fragments: 128 KB
    __shared__ _Float16 h0buf[2 * HBUFSZ];       // double-buffered, 16.5 KB
    __shared__ _Float16 h1buf[2 * HBUFSZ];       // double-buffered, 16.5 KB
    __shared__ float    wfcl[HID];

    const int tid  = threadIdx.x;
    const int lane = tid & 31;
    const int wv   = tid >> 5;
    const int b0   = blockIdx.x * MTILE;
    const int g    = lane >> 4;      // half-wave group (selects K sub-run)
    const int c16  = lane & 15;      // row (A) / column-within-tile (B,C,D)

    // ---------------- one-time setup ----------------
    if (tid < HID) wfcl[tid] = Wfc[tid];

    // hidden state -> parity-0 buffers (f16)
    for (int i = tid; i < MTILE * HID; i += 256) {
        int row = i >> 8, n = i & 255;
        h0buf[row * HSTRIDE + n] = (_Float16)hidden[(size_t)(b0 + row) * HID + n];
        h1buf[row * HSTRIDE + n] = (_Float16)hidden[(size_t)BATCH * HID + (size_t)(b0 + row) * HID + n];
    }
    // swizzle W_ih1 (HIDxHID) into WMMA B-fragment order in LDS
    for (int i = tid; i < 16 * 8 * 32 * 16; i += 256) {
        int e  = i & 15;
        int ln = (i >> 4) & 31;
        int kc = (i >> 9) & 7;
        int nt = i >> 12;
        int K    = 32 * kc + ((e >> 3) << 4) + ((ln >> 4) << 3) + (e & 7);
        int rowW = 16 * nt + (ln & 15);
        Wsw[i] = (_Float16)Wih1[rowW * HID + K];
    }
    // per-wave register-resident B fragments: W_ih0, W_hh0, W_hh1 (wave owns cols 32*wv..32*wv+31)
    v16h bih0f[2][2], bhh0f[2][8], bhh1f[2][8];
    float cb0[2], cb1[2];
#pragma unroll
    for (int j = 0; j < 2; ++j) {
        const int rowW = 16 * (2 * wv + j) + c16;
        cb0[j] = bih0[rowW] + bhh0[rowW];
        cb1[j] = bih1[rowW] + bhh1[rowW];
#pragma unroll
        for (int kc = 0; kc < 2; ++kc) {
            v16h v;
#pragma unroll
            for (int e = 0; e < 16; ++e) {
                int K = 32 * kc + ((e >> 3) << 4) + (g << 3) + (e & 7);
                v[e] = (_Float16)Wih0[rowW * IN_DIM + K];
            }
            bih0f[j][kc] = v;
        }
#pragma unroll
        for (int kc = 0; kc < 8; ++kc) {
            v16h v0, v1;
#pragma unroll
            for (int e = 0; e < 16; ++e) {
                int K = 32 * kc + ((e >> 3) << 4) + (g << 3) + (e & 7);
                v0[e] = (_Float16)Whh0[rowW * HID + K];
                v1[e] = (_Float16)Whh1[rowW * HID + K];
            }
            bhh0f[j][kc] = v0;
            bhh1f[j][kc] = v1;
        }
    }

    const bool ownsOut = (b0 == 496);        // block owning batch row 511
    const float* xbase = x + (size_t)(b0 + c16) * IN_DIM + 8 * g;

    // prefetch x fragments for t = 0 (held in registers across the iteration)
    float4 xr0, xr1, xr2, xr3, xr4, xr5, xr6, xr7;
    {
        const float* px = xbase;             // t = 0
        xr0 = *(const float4*)(px +  0); xr1 = *(const float4*)(px +  4);
        xr2 = *(const float4*)(px + 16); xr3 = *(const float4*)(px + 20);
        xr4 = *(const float4*)(px + 32); xr5 = *(const float4*)(px + 36);
        xr6 = *(const float4*)(px + 48); xr7 = *(const float4*)(px + 52);
    }
    __syncthreads();

    // ---------------- recurrence over time ----------------
    for (int t = 0; t < SEQ; ++t) {
        const int p = t & 1;                 // read parity
        const int q = p ^ 1;                 // write parity
        const _Float16* h0r = h0buf + p * HBUFSZ;
        _Float16*       h0w = h0buf + q * HBUFSZ;
        const _Float16* h1r = h1buf + p * HBUFSZ;
        _Float16*       h1w = h1buf + q * HBUFSZ;

        // ===== layer 0 : h0 = tanh(x_t W_ih0^T + h0 W_hh0^T + bias) =====
        v8f acc[2];
#pragma unroll
        for (int j = 0; j < 2; ++j)
#pragma unroll
            for (int r = 0; r < 8; ++r) acc[j][r] = cb0[j];

        // consume the prefetched x fragments, then immediately prefetch t+1
        v16h ax0 = afrag_from_x(xr0, xr1, xr2, xr3);
        v16h ax1 = afrag_from_x(xr4, xr5, xr6, xr7);
        {
            const int tn = (t + 1 < SEQ) ? (t + 1) : t;
            const float* px = xbase + (size_t)tn * BATCH * IN_DIM;
            xr0 = *(const float4*)(px +  0); xr1 = *(const float4*)(px +  4);
            xr2 = *(const float4*)(px + 16); xr3 = *(const float4*)(px + 20);
            xr4 = *(const float4*)(px + 32); xr5 = *(const float4*)(px + 36);
            xr6 = *(const float4*)(px + 48); xr7 = *(const float4*)(px + 52);
        }
        acc[0] = wmma_f16(ax0, bih0f[0][0], acc[0]);
        acc[1] = wmma_f16(ax0, bih0f[1][0], acc[1]);
        acc[0] = wmma_f16(ax1, bih0f[0][1], acc[0]);
        acc[1] = wmma_f16(ax1, bih0f[1][1], acc[1]);

        // recurrent term from h0 (read parity): software-pipelined A fragments
        {
            v16h a_cur = afrag_lds(h0r, c16, 0, g);
#pragma unroll
            for (int kc = 0; kc < 8; ++kc) {
                v16h a_nxt;
                if (kc < 7) a_nxt = afrag_lds(h0r, c16, kc + 1, g);
                acc[0] = wmma_f16(a_cur, bhh0f[0][kc], acc[0]);
                acc[1] = wmma_f16(a_cur, bhh0f[1][kc], acc[1]);
                a_cur = a_nxt;
            }
        }
        // write new h0 to the other parity (its last readers were fenced last step)
#pragma unroll
        for (int j = 0; j < 2; ++j) {
            const int col = 16 * (2 * wv + j) + c16;
#pragma unroll
            for (int r = 0; r < 8; ++r)
                h0w[(r + 8 * g) * HSTRIDE + col] = (_Float16)tanhf(acc[j][r]);
        }
        __syncthreads();               // new h0 visible to all waves

        // ===== layer 1 : h1 = tanh(h0_new W_ih1^T + h1 W_hh1^T + bias) =====
#pragma unroll
        for (int j = 0; j < 2; ++j)
#pragma unroll
            for (int r = 0; r < 8; ++r) acc[j][r] = cb1[j];

        // software-pipelined: a0/bt0/bt1/a1 fragments loaded one kc ahead
        {
            v16h a0c  = afrag_lds(h0w, c16, 0, g);
            v16h bt0c = bfrag_lds(Wsw, 2 * wv,     0, lane);
            v16h bt1c = bfrag_lds(Wsw, 2 * wv + 1, 0, lane);
            v16h a1c  = afrag_lds(h1r, c16, 0, g);
#pragma unroll
            for (int kc = 0; kc < 8; ++kc) {
                v16h a0n, bt0n, bt1n, a1n;
                if (kc < 7) {
                    a0n  = afrag_lds(h0w, c16, kc + 1, g);
                    bt0n = bfrag_lds(Wsw, 2 * wv,     kc + 1, lane);
                    bt1n = bfrag_lds(Wsw, 2 * wv + 1, kc + 1, lane);
                    a1n  = afrag_lds(h1r, c16, kc + 1, g);
                }
                acc[0] = wmma_f16(a0c, bt0c, acc[0]);
                acc[1] = wmma_f16(a0c, bt1c, acc[1]);
                acc[0] = wmma_f16(a1c, bhh1f[0][kc], acc[0]);
                acc[1] = wmma_f16(a1c, bhh1f[1][kc], acc[1]);
                a0c = a0n; bt0c = bt0n; bt1c = bt1n; a1c = a1n;
            }
        }
#pragma unroll
        for (int j = 0; j < 2; ++j) {
            const int col = 16 * (2 * wv + j) + c16;
#pragma unroll
            for (int r = 0; r < 8; ++r)
                h1w[(r + 8 * g) * HSTRIDE + col] = (_Float16)tanhf(acc[j][r]);
        }
        __syncthreads();               // new h1 visible; safe for tap + next step

        // ===== tap batch row 511 every timestep (reference slices out[:, 511]) =====
        if (ownsOut && wv == 0) {
            float partial = 0.f;
            const _Float16* hp = h1w + 15 * HSTRIDE + lane * 8;
#pragma unroll
            for (int jj = 0; jj < 8; ++jj)
                partial += (float)hp[jj] * wfcl[lane * 8 + jj];
#pragma unroll
            for (int m = 16; m >= 1; m >>= 1)
                partial += __shfl_xor(partial, m, 32);
            if (lane == 0)
                out[t] = 1.0f / (1.0f + expf(-(partial + bfc[0])));
        }
    }
}

extern "C" void kernel_launch(void* const* d_in, const int* in_sizes, int n_in,
                              void* d_out, int out_size, void* d_ws, size_t ws_size,
                              hipStream_t stream) {
    const float* x     = (const float*)d_in[0];
    const float* hid   = (const float*)d_in[1];
    const float* Wih0  = (const float*)d_in[2];
    const float* Whh0  = (const float*)d_in[3];
    const float* bih0v = (const float*)d_in[4];
    const float* bhh0v = (const float*)d_in[5];
    const float* Wih1  = (const float*)d_in[6];
    const float* Whh1  = (const float*)d_in[7];
    const float* bih1v = (const float*)d_in[8];
    const float* bhh1v = (const float*)d_in[9];
    const float* Wfc   = (const float*)d_in[10];
    const float* bfcv  = (const float*)d_in[11];

    rnn_fused<<<BATCH / MTILE, 256, 0, stream>>>(
        x, hid, Wih0, Whh0, bih0v, bhh0v, Wih1, Whh1, bih1v, bhh1v, Wfc, bfcv,
        (float*)d_out);
}
